// ModifiedEncoderLayer_60112362275489
// MI455X (gfx1250) — matrix operand order repaired and checked by reading, not compiled, not run on hardware
//
#include <hip/hip_runtime.h>
#include <math.h>

// ---------------------------------------------------------------------------
// ModifiedEncoderLayer for MI455X (gfx1250): bf16 WMMA GEMMs + flash attention
// A tiles: async global->LDS (ASYNCcnt). W/K tiles: TDM tensor_load (TENSORcnt)
// ---------------------------------------------------------------------------
#define B_    2
#define LSEQ  2048
#define DM    512
#define NH    8
#define DH    64
#define DFF   2048
#define SK    40      // sample_k = min(5*ceil(ln 2048), 2048)
#define NT    40      // n_top
#define MT    (B_ * LSEQ)   // 4096 rows

typedef __attribute__((ext_vector_type(16))) __bf16    v16bf;
typedef __attribute__((ext_vector_type(8)))  float     v8f;
typedef __attribute__((ext_vector_type(4)))  int       v4i;
typedef __attribute__((ext_vector_type(4)))  unsigned  v4u;
typedef __attribute__((ext_vector_type(8)))  int       v8i;

union BF16x16 { v16bf v; v4i q[2]; };

__device__ inline v8f wmma_bf16(const BF16x16& a, const BF16x16& b, v8f c) {
  return __builtin_amdgcn_wmma_f32_16x16x32_bf16(false, a.v, false, b.v,
                                                 (short)0, c, false, false);
}

// LDS byte offset of a __shared__ object: generic address low 32 bits
__device__ inline unsigned lds_off(const void* p) {
  return (unsigned)(uintptr_t)p;
}

// 16B async copy global -> LDS (tracked by ASYNCcnt)
__device__ inline void async_ld_b128(const void* gptr, unsigned ldsaddr) {
  asm volatile("global_load_async_to_lds_b128 %0, %1, off"
               :: "v"(ldsaddr), "v"(gptr) : "memory");
}
__device__ inline void wait_async0() {
  asm volatile("s_wait_asynccnt 0" ::: "memory");
}

// TDM 2D tile load: global (row-major, stride0 elements of 2B) -> LDS,
// with optional per-row LDS padding. Issued once per calling wave.
// D# packing per CDNA5 ISA 8.3/8.4: group0 = {count/type/lds/global addr},
// group1 = {data_size, pad cfg, tensor dims, tile dims, stride}.
__device__ inline void tdm_load_2d(const void* gbase, unsigned ldsaddr,
                                   unsigned td0, unsigned td1,
                                   unsigned tile0, unsigned tile1,
                                   unsigned stride0,
                                   unsigned pad_interval_code,
                                   unsigned pad_amount_code) {
  unsigned long long ga = (unsigned long long)(uintptr_t)gbase;
  v4u g0;
  g0[0] = 1u;                                          // count=1 (valid user D#)
  g0[1] = ldsaddr;                                     // lds_addr [63:32]
  g0[2] = (unsigned)(ga & 0xffffffffu);                // global_addr[31:0]
  g0[3] = (unsigned)((ga >> 32) & 0x01ffffffu)         // global_addr[56:32]
          | (2u << 30);                                // type=2 ("image")
  v8i g1;
  g1[0] = (int)((1u << 16)                             // data_size = 2 bytes
                | (1u << 20)                           // pad_enable
                | (pad_interval_code << 22)
                | (pad_amount_code << 25));
  g1[1] = (int)((td0 & 0xffffu) << 16);                // tensor_dim0[15:0]
  g1[2] = (int)((td0 >> 16) | ((td1 & 0xffffu) << 16));// dim0 hi | dim1 lo
  g1[3] = (int)((td1 >> 16) | (tile0 << 16));          // dim1 hi | tile_dim0
  g1[4] = (int)(tile1 & 0xffffu);                      // tile_dim1 (tile_dim2=0)
  g1[5] = (int)stride0;                                // tensor_dim0_stride lo
  g1[6] = 0;                                           // stride0 hi | dim1_stride lo
  g1[7] = 0;
  v4i z4 = {0, 0, 0, 0};
  v8i z8 = {0, 0, 0, 0, 0, 0, 0, 0};
  __builtin_amdgcn_tensor_load_to_lds(g0, g1, z4, z4, z8, 0);
}
__device__ inline void wait_tensor0() {
  __builtin_amdgcn_s_wait_tensorcnt((short)0);
}

// ---------------------------------------------------------------------------
// fp32 -> bf16 convert
// ---------------------------------------------------------------------------
__global__ __launch_bounds__(256) void f2bf_k(const float* __restrict__ in,
                                              __bf16* __restrict__ out, int n) {
  int i = blockIdx.x * 256 + threadIdx.x;
  if (i < n) out[i] = (__bf16)in[i];
}

// ---------------------------------------------------------------------------
// GEMM: C[M,N] = A[M,K] (bf16) @ W[N,K]^T (bf16) + bias, opt relu, fp32/bf16 out
// block tile 128(M) x 64(N), K-step 64, 8 waves.
// A tile: per-lane async loads. W tile: one TDM descriptor (wave 0).
// ---------------------------------------------------------------------------
__global__ __launch_bounds__(256) void gemm_bf16_k(
    const __bf16* __restrict__ A, const __bf16* __restrict__ W,
    const float* __restrict__ bias, float* __restrict__ Cf,
    __bf16* __restrict__ Cb, int M, int N, int K, int relu) {
  __shared__ __align__(16) __bf16 As[128][72];   // 128x64 + pad(8)
  __shared__ __align__(16) __bf16 Ws[64][72];    // 64x64 + pad(8) (TDM-padded)
  const int tid = threadIdx.x, lane = tid & 31, wave = tid >> 5;
  const int khalf = lane >> 4, ln = lane & 15, c0 = khalf * 8;
  const int m0 = blockIdx.y * 128, n0 = blockIdx.x * 64;
  const v8f vzero = {0.f, 0.f, 0.f, 0.f, 0.f, 0.f, 0.f, 0.f};
  v8f acc[4];
#pragma unroll
  for (int t = 0; t < 4; t++) acc[t] = vzero;

  for (int kb = 0; kb < K; kb += 64) {
    // stage A tile 128x64 bf16: 1024 16B-chunks, 4 per thread (async)
#pragma unroll
    for (int c = 0; c < 4; c++) {
      int ch = tid + c * 256;
      int row = ch >> 3, kc = (ch & 7) * 8;
      async_ld_b128(A + (size_t)(m0 + row) * K + kb + kc,
                    lds_off(&As[row][kc]));
    }
    // stage W tile 64x64 bf16 with one TDM 2D descriptor (row pad = 8 bf16):
    // pad_interval: 32 DWORDs (code 4) = one 128B row; pad_amount: 4 DWORDs (code 3)
    if (wave == 0)
      tdm_load_2d(W + (size_t)n0 * K + kb, lds_off(&Ws[0][0]),
                  (unsigned)(K - kb), (unsigned)(N - n0), 64u, 64u,
                  (unsigned)K, 4u, 3u);
    if (kb + 64 < K)
      __builtin_prefetch(A + (size_t)(m0 + (tid >> 1)) * K + kb + 64 + (tid & 1) * 32, 0, 0);
    wait_async0();
    if (wave == 0) wait_tensor0();
    __syncthreads();

#pragma unroll
    for (int kk = 0; kk < 2; kk++) {
      // batch all fragment loads, then issue the 4 WMMAs
      BF16x16 af, bf[4];
      af.q[0] = *(const v4i*)&As[wave * 16 + ln][kk * 32 + khalf * 16];
      af.q[1] = *(const v4i*)&As[wave * 16 + ln][kk * 32 + khalf * 16 + 8];
#pragma unroll
      for (int t = 0; t < 4; t++) {
        bf[t].q[0] = *(const v4i*)&Ws[t * 16 + ln][kk * 32 + khalf * 16];
        bf[t].q[1] = *(const v4i*)&Ws[t * 16 + ln][kk * 32 + khalf * 16 + 8];
      }
#pragma unroll
      for (int t = 0; t < 4; t++) acc[t] = wmma_bf16(af, bf[t], acc[t]);
    }
    __syncthreads();
  }

#pragma unroll
  for (int t = 0; t < 4; t++) {
#pragma unroll
    for (int j = 0; j < 8; j++) {
      int m = m0 + wave * 16 + j + khalf * 8;
      int n = n0 + t * 16 + ln;
      float v = acc[t][j];
      if (bias) v += bias[n];
      if (relu) v = fmaxf(v, 0.f);
      size_t off = (size_t)m * N + n;
      if (Cf) Cf[off] = v;
      if (Cb) Cb[off] = (__bf16)v;
    }
  }
}

// ---------------------------------------------------------------------------
// Flash attention (full branch). Q/K/V bf16 [B*L, 512] head-major, per-head 64.
// grid: (L/128, B*H). 8 waves, each owns a 16-query tile; 32-key steps.
// K tile: TDM 2D load (wave 0). V tile: register transpose through LDS.
// ---------------------------------------------------------------------------
__global__ __launch_bounds__(256) void flash_attn_k(
    const __bf16* __restrict__ Qb, const __bf16* __restrict__ Kb,
    const __bf16* __restrict__ Vb, float* __restrict__ ctxF,
    __bf16* __restrict__ ctxB) {
  __shared__ __align__(16) __bf16 Ks[32][80];    // [key][d]  contiguous d (TDM pad 16)
  __shared__ __align__(16) __bf16 Vst[64][40];   // [d][key]  contiguous key
  __shared__ __align__(16) __bf16 Ps[8][16][40]; // per-wave P tile [row][key]

  const int tid = threadIdx.x, lane = tid & 31, wave = tid >> 5;
  const int khalf = lane >> 4, ln = lane & 15, c0 = khalf * 8;
  const int bh = blockIdx.y, b = bh >> 3, h = bh & 7;
  const int q0 = blockIdx.x * 128 + wave * 16;
  const v8f vzero = {0.f, 0.f, 0.f, 0.f, 0.f, 0.f, 0.f, 0.f};

  // A-fragments of Q (16 queries x 64 d -> two 16x32 fragments), loaded once
  BF16x16 qf[2];
  {
    const __bf16* qrow = Qb + ((size_t)(b * LSEQ + q0 + ln)) * DM + h * DH;
#pragma unroll
    for (int kk = 0; kk < 2; kk++) {
      qf[kk].q[0] = *(const v4i*)(qrow + kk * 32 + khalf * 16);
      qf[kk].q[1] = *(const v4i*)(qrow + kk * 32 + khalf * 16 + 8);
    }
  }

  float mrow[8], lrow[8];
  v8f accO[4];
#pragma unroll
  for (int j = 0; j < 8; j++) { mrow[j] = -INFINITY; lrow[j] = 0.f; }
#pragma unroll
  for (int t = 0; t < 4; t++) accO[t] = vzero;

  for (int k0 = 0; k0 < LSEQ; k0 += 32) {
    // K tile 32x64 via TDM (row pad = 16 bf16 = 8 DWORDs -> code 7)
    if (wave == 0)
      tdm_load_2d(Kb + ((size_t)(b * LSEQ + k0)) * DM + h * DH,
                  lds_off(&Ks[0][0]),
                  (unsigned)DM, (unsigned)(LSEQ - k0), 64u, 32u,
                  (unsigned)DM, 4u, 7u);
    // V tile: per-lane load + transpose into LDS
    {
      int key = tid >> 3, d0 = (tid & 7) * 8;
      size_t goff = ((size_t)(b * LSEQ + k0 + key)) * DM + h * DH + d0;
      v4i vq = *(const v4i*)(Vb + goff);
      const __bf16* vp = (const __bf16*)&vq;
#pragma unroll
      for (int e = 0; e < 8; e++) Vst[d0 + e][key] = vp[e];
    }
    if (wave == 0) wait_tensor0();
    __syncthreads();

    // S = Q K^T : two 16x16 tiles (keys k0..+15, k0+16..+31), K-dim = d (2x32)
    v8f s0 = vzero, s1 = vzero;
#pragma unroll
    for (int kk = 0; kk < 2; kk++) {
      BF16x16 kf0, kf1;
      kf0.q[0] = *(const v4i*)&Ks[ln][kk * 32 + khalf * 16];
      kf0.q[1] = *(const v4i*)&Ks[ln][kk * 32 + khalf * 16 + 8];
      kf1.q[0] = *(const v4i*)&Ks[16 + ln][kk * 32 + khalf * 16];
      kf1.q[1] = *(const v4i*)&Ks[16 + ln][kk * 32 + khalf * 16 + 8];
      s0 = wmma_bf16(qf[kk], kf0, s0);
      s1 = wmma_bf16(qf[kk], kf1, s1);
    }

    // online softmax (per-row stats via 16-lane xor reductions)
#pragma unroll
    for (int j = 0; j < 8; j++) {
      float a = s0[j] * 0.125f, bb = s1[j] * 0.125f;
      float mx = fmaxf(a, bb);
#pragma unroll
      for (int o = 1; o < 16; o <<= 1) mx = fmaxf(mx, __shfl_xor(mx, o, 32));
      float nm = fmaxf(mrow[j], mx);
      float corr = __expf(mrow[j] - nm);
      mrow[j] = nm;
      float p0 = __expf(a - nm), p1 = __expf(bb - nm);
      float rs = p0 + p1;
#pragma unroll
      for (int o = 1; o < 16; o <<= 1) rs += __shfl_xor(rs, o, 32);
      lrow[j] = lrow[j] * corr + rs;
#pragma unroll
      for (int t = 0; t < 4; t++) accO[t][j] *= corr;
      int r = j + khalf * 8;
      Ps[wave][r][ln] = (__bf16)p0;
      Ps[wave][r][16 + ln] = (__bf16)p1;
    }
    asm volatile("s_wait_dscnt 0" ::: "memory");

    // O += P @ V  (A = P 16x32, B = V 32x16 per 16-d subtile)
    BF16x16 pa, vf[4];
    pa.q[0] = *(const v4i*)&Ps[wave][ln][khalf * 16];
    pa.q[1] = *(const v4i*)&Ps[wave][ln][khalf * 16 + 8];
#pragma unroll
    for (int dt = 0; dt < 4; dt++) {
      vf[dt].q[0] = *(const v4i*)&Vst[dt * 16 + ln][khalf * 16];
      vf[dt].q[1] = *(const v4i*)&Vst[dt * 16 + ln][khalf * 16 + 8];
    }
#pragma unroll
    for (int dt = 0; dt < 4; dt++) accO[dt] = wmma_bf16(pa, vf[dt], accO[dt]);
    __syncthreads();
  }

#pragma unroll
  for (int dt = 0; dt < 4; dt++) {
#pragma unroll
    for (int j = 0; j < 8; j++) {
      int q = q0 + j + khalf * 8;
      int col = h * DH + dt * 16 + ln;
      float o = accO[dt][j] / lrow[j];
      size_t off = ((size_t)(b * LSEQ + q)) * DM + col;
      if (ctxF) ctxF[off] = o;
      ctxB[off] = (__bf16)o;
    }
  }
}

// ---------------------------------------------------------------------------
// Prob attention pieces (small; fp32 VALU)
// ---------------------------------------------------------------------------
__global__ __launch_bounds__(256) void qk_sample_k(
    const float* __restrict__ q, const float* __restrict__ k,
    const int* __restrict__ idx, float* __restrict__ qks) {
  int t = blockIdx.x * 256 + threadIdx.x;            // over B*H*L*SK
  if (t >= B_ * NH * LSEQ * SK) return;
  int j = t % SK, l = (t / SK) % LSEQ, h = (t / (SK * LSEQ)) % NH,
      b = t / (SK * LSEQ * NH);
  int ks = idx[l * SK + j];
  const float* qp = q + ((size_t)(b * LSEQ + l)) * DM + h * DH;
  const float* kp = k + ((size_t)(b * LSEQ + ks)) * DM + h * DH;
  float s = 0.f;
#pragma unroll 8
  for (int d = 0; d < DH; d++) s += qp[d] * kp[d];
  qks[((size_t)((b * NH + h) * LSEQ + l)) * SK + j] = s;
}

__global__ __launch_bounds__(256) void m_metric_k(const float* __restrict__ qks,
                                                  float* __restrict__ Mg) {
  int i = blockIdx.x * 256 + threadIdx.x;            // over B*H*L
  if (i >= B_ * NH * LSEQ) return;
  const float* r = qks + (size_t)i * SK;
  float mx = -INFINITY, s = 0.f;
  for (int j = 0; j < SK; j++) { float v = r[j]; mx = fmaxf(mx, v); s += v; }
  Mg[i] = mx - s * (1.f / (float)LSEQ);
}

__global__ __launch_bounds__(256) void topk_k(const float* __restrict__ Mg,
                                              int* __restrict__ mtop) {
  int bh = blockIdx.x, tid = threadIdx.x;
  __shared__ float Ml[LSEQ];
  __shared__ float rv[256];
  __shared__ int ri[256];
  for (int i = tid; i < LSEQ; i += 256) Ml[i] = Mg[(size_t)bh * LSEQ + i];
  __syncthreads();
  for (int u = 0; u < NT; u++) {
    float bv = -INFINITY; int bi = 0;
    for (int i = tid; i < LSEQ; i += 256) {
      float v = Ml[i];
      if (v > bv) { bv = v; bi = i; }
    }
    rv[tid] = bv; ri[tid] = bi; __syncthreads();
    for (int s = 128; s > 0; s >>= 1) {
      if (tid < s && rv[tid + s] > rv[tid]) { rv[tid] = rv[tid + s]; ri[tid] = ri[tid + s]; }
      __syncthreads();
    }
    if (tid == 0) { mtop[bh * NT + u] = ri[0]; Ml[ri[0]] = -INFINITY; }
    __syncthreads();
  }
}

__global__ __launch_bounds__(256) void prob_row_k(
    const float* __restrict__ q2, const float* __restrict__ k2,
    const float* __restrict__ v2, const int* __restrict__ mtop,
    float* __restrict__ upd) {
  int blk = blockIdx.x, bh = blk / NT, u = blk % NT;
  int b = bh >> 3, h = bh & 7, tid = threadIdx.x;
  __shared__ float srow[LSEQ];
  __shared__ float qs[DH];
  __shared__ float red[256];
  __shared__ float a4[4][DH];
  int qi = mtop[bh * NT + u];
  if (tid < DH) qs[tid] = q2[((size_t)(b * LSEQ + qi)) * DM + h * DH + tid];
  __syncthreads();
  for (int key = tid; key < LSEQ; key += 256) {
    const float* kp = k2 + ((size_t)(b * LSEQ + key)) * DM + h * DH;
    float s = 0.f;
#pragma unroll 8
    for (int d = 0; d < DH; d++) s += qs[d] * kp[d];
    srow[key] = s * 0.125f;
  }
  __syncthreads();
  float mx = -INFINITY;
  for (int key = tid; key < LSEQ; key += 256) mx = fmaxf(mx, srow[key]);
  red[tid] = mx; __syncthreads();
  for (int s = 128; s > 0; s >>= 1) { if (tid < s) red[tid] = fmaxf(red[tid], red[tid + s]); __syncthreads(); }
  mx = red[0]; __syncthreads();
  float psum = 0.f;
  for (int key = tid; key < LSEQ; key += 256) {
    float p = __expf(srow[key] - mx);
    srow[key] = p; psum += p;
  }
  red[tid] = psum; __syncthreads();
  for (int s = 128; s > 0; s >>= 1) { if (tid < s) red[tid] += red[tid + s]; __syncthreads(); }
  float denom = red[0]; __syncthreads();
  int d = tid & 63, sl = tid >> 6;
  float part = 0.f;
  for (int key = sl; key < LSEQ; key += 4)
    part += srow[key] * v2[((size_t)(b * LSEQ + key)) * DM + h * DH + d];
  a4[sl][d] = part; __syncthreads();
  if (tid < DH)
    upd[((size_t)bh * NT + u) * DH + tid] =
        (a4[0][tid] + a4[1][tid] + a4[2][tid] + a4[3][tid]) / denom;
}

__global__ __launch_bounds__(256) void vmean_k(const float* __restrict__ v2,
                                               float* __restrict__ vmean) {
  int bh = blockIdx.x, b = bh >> 3, h = bh & 7, tid = threadIdx.x;
  __shared__ float a4[4][DH];
  int d = tid & 63, sl = tid >> 6;
  float s = 0.f;
  for (int l = sl; l < LSEQ; l += 4)
    s += v2[((size_t)(b * LSEQ + l)) * DM + h * DH + d];
  a4[sl][d] = s; __syncthreads();
  if (tid < DH)
    vmean[bh * DH + tid] =
        (a4[0][tid] + a4[1][tid] + a4[2][tid] + a4[3][tid]) * (1.f / (float)LSEQ);
}

__global__ __launch_bounds__(256) void ctx_fill_k(const float* __restrict__ vmean,
                                                  float* __restrict__ ctxf) {
  int i = blockIdx.x * 256 + threadIdx.x;            // over B*L*DM
  if (i >= MT * DM) return;
  int b = i >> 20;                                    // L*DM = 2^20
  ctxf[i] = vmean[b * DM + (i & (DM - 1))];
}

__global__ __launch_bounds__(256) void scatter_k(const int* __restrict__ mtop,
                                                 const float* __restrict__ upd,
                                                 float* __restrict__ ctxf) {
  int i = blockIdx.x * 256 + threadIdx.x;            // over 16*NT*DH
  if (i >= B_ * NH * NT * DH) return;
  int d = i & 63, u = (i >> 6) % NT, bh = i / (NT * DH);
  int b = bh >> 3, h = bh & 7;
  int qi = mtop[bh * NT + u];
  ctxf[((size_t)(b * LSEQ + qi)) * DM + h * DH + d] = upd[i];
}

// ---------------------------------------------------------------------------
// gate combine + LN1 ; residual + LN2
// ---------------------------------------------------------------------------
__global__ __launch_bounds__(256) void combine_ln1_k(
    const float* __restrict__ x, const float* __restrict__ a1,
    const float* __restrict__ a2, const float* __restrict__ gate,
    const float* __restrict__ gg, const float* __restrict__ gb,
    float* __restrict__ x1f, __bf16* __restrict__ x1b) {
  __shared__ float red[256];
  int row = blockIdx.x, tid = threadIdx.x;
  size_t base = (size_t)row * DM;
  float g = 1.f / (1.f + __expf(-gate[0]));
  float v0 = x[base + tid] + g * a1[base + tid] + (1.f - g) * a2[base + tid];
  float v1 = x[base + tid + 256] + g * a1[base + tid + 256] + (1.f - g) * a2[base + tid + 256];
  red[tid] = v0 + v1; __syncthreads();
  for (int s = 128; s > 0; s >>= 1) { if (tid < s) red[tid] += red[tid + s]; __syncthreads(); }
  float mean = red[0] * (1.f / (float)DM); __syncthreads();
  float d0 = v0 - mean, d1 = v1 - mean;
  red[tid] = d0 * d0 + d1 * d1; __syncthreads();
  for (int s = 128; s > 0; s >>= 1) { if (tid < s) red[tid] += red[tid + s]; __syncthreads(); }
  float rs = rsqrtf(red[0] * (1.f / (float)DM) + 1e-5f); __syncthreads();
  float o0 = d0 * rs * gg[tid] + gb[tid];
  float o1 = d1 * rs * gg[tid + 256] + gb[tid + 256];
  x1f[base + tid] = o0;           x1b[base + tid] = (__bf16)o0;
  x1f[base + tid + 256] = o1;     x1b[base + tid + 256] = (__bf16)o1;
}

__global__ __launch_bounds__(256) void add_ln2_k(
    const float* __restrict__ x1, const float* __restrict__ ff,
    const float* __restrict__ gg, const float* __restrict__ gb,
    float* __restrict__ out) {
  __shared__ float red[256];
  int row = blockIdx.x, tid = threadIdx.x;
  size_t base = (size_t)row * DM;
  float v0 = x1[base + tid] + ff[base + tid];
  float v1 = x1[base + tid + 256] + ff[base + tid + 256];
  red[tid] = v0 + v1; __syncthreads();
  for (int s = 128; s > 0; s >>= 1) { if (tid < s) red[tid] += red[tid + s]; __syncthreads(); }
  float mean = red[0] * (1.f / (float)DM); __syncthreads();
  float d0 = v0 - mean, d1 = v1 - mean;
  red[tid] = d0 * d0 + d1 * d1; __syncthreads();
  for (int s = 128; s > 0; s >>= 1) { if (tid < s) red[tid] += red[tid + s]; __syncthreads(); }
  float rs = rsqrtf(red[0] * (1.f / (float)DM) + 1e-5f); __syncthreads();
  out[base + tid]       = d0 * rs * gg[tid] + gb[tid];
  out[base + tid + 256] = d1 * rs * gg[tid + 256] + gb[tid + 256];
}

// ---------------------------------------------------------------------------
// launch
// ---------------------------------------------------------------------------
extern "C" void kernel_launch(void* const* d_in, const int* in_sizes, int n_in,
                              void* d_out, int out_size, void* d_ws, size_t ws_size,
                              hipStream_t stream) {
  const float* x   = (const float*)d_in[0];
  const float* q1w = (const float*)d_in[1];  const float* q1b = (const float*)d_in[2];
  const float* k1w = (const float*)d_in[3];  const float* k1b = (const float*)d_in[4];
  const float* v1w = (const float*)d_in[5];  const float* v1b = (const float*)d_in[6];
  const float* o1w = (const float*)d_in[7];  const float* o1b = (const float*)d_in[8];
  const float* q2w = (const float*)d_in[9];  const float* q2b = (const float*)d_in[10];
  const float* k2w = (const float*)d_in[11]; const float* k2b = (const float*)d_in[12];
  const float* v2w = (const float*)d_in[13]; const float* v2b = (const float*)d_in[14];
  const float* o2w = (const float*)d_in[15]; const float* o2b = (const float*)d_in[16];
  const float* f1w = (const float*)d_in[17]; const float* f1b = (const float*)d_in[18];
  const float* f2w = (const float*)d_in[19]; const float* f2b = (const float*)d_in[20];
  const float* ln1g = (const float*)d_in[21]; const float* ln1b = (const float*)d_in[22];
  const float* ln2g = (const float*)d_in[23]; const float* ln2b = (const float*)d_in[24];
  const float* gate = (const float*)d_in[25];
  const int*   idxs = (const int*)d_in[26];

  char* p = (char*)d_ws;
  auto alloc = [&](size_t nbytes) -> char* {
    char* r = p; p += (nbytes + 255) & ~(size_t)255; return r;
  };

  __bf16* xbf  = (__bf16*)alloc((size_t)MT * DM * 2);
  __bf16* wq1  = (__bf16*)alloc((size_t)DM * DM * 2);
  __bf16* wk1  = (__bf16*)alloc((size_t)DM * DM * 2);
  __bf16* wv1  = (__bf16*)alloc((size_t)DM * DM * 2);
  __bf16* wo1  = (__bf16*)alloc((size_t)DM * DM * 2);
  __bf16* wq2  = (__bf16*)alloc((size_t)DM * DM * 2);
  __bf16* wk2  = (__bf16*)alloc((size_t)DM * DM * 2);
  __bf16* wv2  = (__bf16*)alloc((size_t)DM * DM * 2);
  __bf16* wo2  = (__bf16*)alloc((size_t)DM * DM * 2);
  __bf16* wf1  = (__bf16*)alloc((size_t)DFF * DM * 2);
  __bf16* wf2  = (__bf16*)alloc((size_t)DM * DFF * 2);
  __bf16* Q1b  = (__bf16*)alloc((size_t)MT * DM * 2);
  __bf16* K1b  = (__bf16*)alloc((size_t)MT * DM * 2);
  __bf16* V1b  = (__bf16*)alloc((size_t)MT * DM * 2);
  __bf16* ctx1b = (__bf16*)alloc((size_t)MT * DM * 2);
  float*  attn1 = (float*)alloc((size_t)MT * DM * 4);
  float*  q2f  = (float*)alloc((size_t)MT * DM * 4);
  float*  k2f  = (float*)alloc((size_t)MT * DM * 4);
  float*  v2f  = (float*)alloc((size_t)MT * DM * 4);
  float*  qks  = (float*)alloc((size_t)B_ * NH * LSEQ * SK * 4);
  float*  Mv   = (float*)alloc((size_t)B_ * NH * LSEQ * 4);
  int*    mtop = (int*)alloc((size_t)B_ * NH * NT * 4);
  float*  upd  = (float*)alloc((size_t)B_ * NH * NT * DH * 4);
  float*  vmn  = (float*)alloc((size_t)B_ * NH * DH * 4);
  float*  ctx2f = (float*)alloc((size_t)MT * DM * 4);
  __bf16* ctx2b = (__bf16*)alloc((size_t)MT * DM * 2);
  float*  attn2 = (float*)alloc((size_t)MT * DM * 4);
  float*  x1f  = (float*)alloc((size_t)MT * DM * 4);
  __bf16* x1b  = (__bf16*)alloc((size_t)MT * DM * 2);
  __bf16* hb   = (__bf16*)alloc((size_t)MT * DFF * 2);
  float*  fff  = (float*)alloc((size_t)MT * DM * 4);

  auto cvt = [&](const float* s, __bf16* d, int n) {
    f2bf_k<<<(n + 255) / 256, 256, 0, stream>>>(s, d, n);
  };
  cvt(x, xbf, MT * DM);
  cvt(q1w, wq1, DM * DM); cvt(k1w, wk1, DM * DM);
  cvt(v1w, wv1, DM * DM); cvt(o1w, wo1, DM * DM);
  cvt(q2w, wq2, DM * DM); cvt(k2w, wk2, DM * DM);
  cvt(v2w, wv2, DM * DM); cvt(o2w, wo2, DM * DM);
  cvt(f1w, wf1, DFF * DM); cvt(f2w, wf2, DM * DFF);

  auto gemm = [&](const __bf16* A, const __bf16* W, const float* bias,
                  float* Cf, __bf16* Cb, int M, int N, int K, int relu) {
    dim3 g(N / 64, M / 128);
    gemm_bf16_k<<<g, 256, 0, stream>>>(A, W, bias, Cf, Cb, M, N, K, relu);
  };

  // full-attention branch
  gemm(xbf, wq1, q1b, nullptr, Q1b, MT, DM, DM, 0);
  gemm(xbf, wk1, k1b, nullptr, K1b, MT, DM, DM, 0);
  gemm(xbf, wv1, v1b, nullptr, V1b, MT, DM, DM, 0);
  flash_attn_k<<<dim3(LSEQ / 128, B_ * NH), 256, 0, stream>>>(Q1b, K1b, V1b,
                                                              nullptr, ctx1b);
  gemm(ctx1b, wo1, o1b, attn1, nullptr, MT, DM, DM, 0);

  // prob-attention branch
  gemm(xbf, wq2, q2b, q2f, nullptr, MT, DM, DM, 0);
  gemm(xbf, wk2, k2b, k2f, nullptr, MT, DM, DM, 0);
  gemm(xbf, wv2, v2b, v2f, nullptr, MT, DM, DM, 0);
  qk_sample_k<<<(B_ * NH * LSEQ * SK) / 256, 256, 0, stream>>>(q2f, k2f, idxs, qks);
  m_metric_k<<<(B_ * NH * LSEQ) / 256, 256, 0, stream>>>(qks, Mv);
  topk_k<<<B_ * NH, 256, 0, stream>>>(Mv, mtop);
  prob_row_k<<<B_ * NH * NT, 256, 0, stream>>>(q2f, k2f, v2f, mtop, upd);
  vmean_k<<<B_ * NH, 256, 0, stream>>>(v2f, vmn);
  ctx_fill_k<<<(MT * DM) / 256, 256, 0, stream>>>(vmn, ctx2f);
  scatter_k<<<(B_ * NH * NT * DH + 255) / 256, 256, 0, stream>>>(mtop, upd, ctx2f);
  cvt(ctx2f, ctx2b, MT * DM);
  gemm(ctx2b, wo2, o2b, attn2, nullptr, MT, DM, DM, 0);

  // gate combine + LN1
  combine_ln1_k<<<MT, 256, 0, stream>>>(x, attn1, attn2, gate, ln1g, ln1b, x1f, x1b);

  // FFN
  gemm(x1b, wf1, f1b, nullptr, hb, MT, DFF, DM, 1);
  gemm(hb, wf2, f2b, fff, nullptr, MT, DM, DFF, 0);

  // residual + LN2 -> output
  add_ln2_k<<<MT, 256, 0, stream>>>(x1f, fff, ln2g, ln2b, (float*)d_out);

  (void)in_sizes; (void)n_in; (void)out_size; (void)ws_size;
}
